// Generator_46797963657235
// MI455X (gfx1250) — compile-verified, matrix operand order checked by reading, not run on hardware
//
#include <hip/hip_runtime.h>
#include <cstdint>

// ---------------------------------------------------------------------------
// CDNA5 (gfx1250) WMMA types
// ---------------------------------------------------------------------------
typedef __attribute__((ext_vector_type(16))) __bf16 v16bf;
typedef __attribute__((ext_vector_type(8)))  float  v8f;

union FragAB { v16bf v; uint32_t u[8]; };
union FragC  { v8f   v; float    f[8]; };

__device__ __forceinline__ uint16_t f2bf(float f) {
  union { float f; uint32_t u; } x; x.f = f;
  uint32_t r = x.u + 0x7FFFu + ((x.u >> 16) & 1u);   // round-to-nearest-even
  return (uint16_t)(r >> 16);
}
__device__ __forceinline__ uint32_t pack_bf2(float lo, float hi) {
  return (uint32_t)f2bf(lo) | ((uint32_t)f2bf(hi) << 16);
}

// 16-bit A/B fragment K-pair map for V_WMMA_*_16X16X32 (wave32):
// lanes 0-15 hold K={0..7,16..23}, lanes 16-31 hold +8 (ISA 7.12.2)
__device__ __forceinline__ int kmap(int v, int khalf) {
  return ((v < 4) ? (2 * v) : (16 + 2 * (v - 4))) + 8 * khalf;
}

// ---------------------------------------------------------------------------
// Weight repack: f32 (OIHW, or IOHW for deconv) -> bf16 [Mpad][Kpad], zero pad
// ---------------------------------------------------------------------------
__global__ void repack_w_kernel(const float* __restrict__ w, uint16_t* __restrict__ A,
                                int M, int K, int Kpad, size_t total,
                                int mode, int Cout, int KHW) {
  size_t idx = (size_t)blockIdx.x * blockDim.x + threadIdx.x;
  if (idx >= total) return;
  int m = (int)(idx / Kpad);
  int k = (int)(idx - (size_t)m * Kpad);
  float val = 0.0f;
  if (m < M && k < K) {
    if (mode == 0) {                       // conv: w[Cout][Cin][KH][KW]
      val = w[(size_t)m * K + k];
    } else {                               // deconv: w[Cin][Cout][KH][KW]
      int c = k / KHW, r = k - c * KHW;
      val = w[((size_t)c * Cout + m) * KHW + r];
    }
  }
  A[idx] = f2bf(val);
}

// ---------------------------------------------------------------------------
// Implicit-GEMM conv / transposed-conv (geometry templated so all the im2col
// div/mod lowers to magic multiplies):
//   C[Cout][B*OH*OW] = A[Cout][Cin*KH*KW] x im2col(X)
// Block = 256 threads (8 waves), 64x64 tile, K-chunks of 32 (bf16).
// Register-staged double buffering: next chunk's global loads are in flight
// while WMMAs consume the current LDS buffer (2 LDS buffers, 1 barrier/iter).
// LDS rows padded to 34 halves (17-bank skew -> conflict-free frag reads).
// ---------------------------------------------------------------------------
#define LDSTR 34

template <int KH, int KW, int STRIDE, int PAD, int MODE, int ACT>
__global__ __launch_bounds__(256) void conv_gemm_wmma(
    const float* __restrict__ x, const uint16_t* __restrict__ A,
    float* __restrict__ y,
    int Bn, int Cin, int H, int W, int Cout, int OH, int OW,
    int Kdim, int Kpad) {
  __shared__ uint16_t Alds[2][64 * LDSTR];
  __shared__ uint16_t Blds[2][64 * LDSTR];

  constexpr int KHW = KH * KW;
  const int Ntot = Bn * OH * OW;
  const int n0 = blockIdx.x * 64;
  const int m0 = blockIdx.y * 64;
  const int t = threadIdx.x;
  const int wv = t >> 5, lane = t & 31;
  const int ml = lane & 15, khalf = lane >> 4;
  const int mtile = wv & 3, ntile0 = (wv >> 2) * 2;
  const int OHW = OH * OW, HWin = H * W;

  const int row8 = t >> 2;            // 0..63
  const int col8 = (t & 3) * 8;       // 0,8,16,24

  // ---- loop-invariant im2col decode for this thread's B row ----
  const int ng = n0 + row8;
  const bool nval = (ng < Ntot);
  int oh = 0, ow = 0;
  const float* xbase = x;
  if (nval) {
    int b = ng / OHW; int prem = ng - b * OHW;
    oh = prem / OW; ow = prem - oh * OW;
    xbase = x + (size_t)b * Cin * HWin;
  }
  const int ih0 = oh * STRIDE - PAD;      // conv mode
  const int iw0 = ow * STRIDE - PAD;

  FragC acc0, acc1;
#pragma unroll
  for (int i = 0; i < 8; ++i) { acc0.f[i] = 0.0f; acc1.f[i] = 0.0f; }

  // staging registers (double set)
  float breg[2][8];
  uint4 areg[2];

  auto load_tile = [&](int k0, int par) {
    areg[par] = *(const uint4*)(A + (size_t)(m0 + row8) * Kpad + (k0 + col8));
#pragma unroll
    for (int j = 0; j < 8; ++j) {
      int k = k0 + col8 + j;
      float val = 0.0f;
      if (nval && k < Kdim) {
        int c = k / KHW, rem = k - c * KHW;       // constexpr divisors
        int kh = rem / KW, kw = rem - kh * KW;
        if (MODE == 0) {
          int ih = ih0 + kh, iw = iw0 + kw;
          if (ih >= 0 && ih < H && iw >= 0 && iw < W)
            val = xbase[(size_t)c * HWin + ih * W + iw];
        } else {                                  // deconv gather (k=4,s=2,p=1)
          int ihn = oh + PAD - kh, iwn = ow + PAD - kw;
          if (ihn >= 0 && iwn >= 0 && !(ihn & 1) && !(iwn & 1)) {
            int ih = ihn >> 1, iw = iwn >> 1;
            if (ih < H && iw < W) val = xbase[(size_t)c * HWin + ih * W + iw];
          }
        }
      }
      breg[par][j] = val;
    }
  };

  auto store_tile = [&](int par) {
    uint32_t* ad = (uint32_t*)&Alds[par][row8 * LDSTR + col8];
    ad[0] = areg[par].x; ad[1] = areg[par].y; ad[2] = areg[par].z; ad[3] = areg[par].w;
    uint32_t* bd = (uint32_t*)&Blds[par][row8 * LDSTR + col8];
#pragma unroll
    for (int j = 0; j < 4; ++j)
      bd[j] = pack_bf2(breg[par][2 * j], breg[par][2 * j + 1]);
  };

  const int nch = Kpad >> 5;
  load_tile(0, 0);
  for (int i = 0; i < nch; ++i) {
    const int par = i & 1;
    store_tile(par);
    if (i + 1 < nch) load_tile((i + 1) * 32, par ^ 1);
    __syncthreads();

    FragAB af, bf0, bf1;
#pragma unroll
    for (int v = 0; v < 8; ++v) {
      int kp = kmap(v, khalf);
      af.u[v]  = *(const uint32_t*)&Alds[par][(mtile * 16 + ml) * LDSTR + kp];
      bf0.u[v] = *(const uint32_t*)&Blds[par][(ntile0 * 16 + ml) * LDSTR + kp];
      bf1.u[v] = *(const uint32_t*)&Blds[par][((ntile0 + 1) * 16 + ml) * LDSTR + kp];
    }
    acc0.v = __builtin_amdgcn_wmma_f32_16x16x32_bf16(false, af.v, false, bf0.v,
                                                     (short)0, acc0.v, false, false);
    acc1.v = __builtin_amdgcn_wmma_f32_16x16x32_bf16(false, af.v, false, bf1.v,
                                                     (short)0, acc1.v, false, false);
  }

  // ---- epilogue: C layout = VGPR r -> row r+8*khalf, lane&15 -> col ----
#pragma unroll
  for (int j = 0; j < 2; ++j) {
    int ngj = n0 + (ntile0 + j) * 16 + ml;
    if (ngj >= Ntot) continue;
    int b = ngj / OHW; int prem = ngj - b * OHW;
    int ohj = prem / OW, owj = prem - ohj * OW;
    float* yb = y + (size_t)b * Cout * OHW + ohj * OW + owj;
#pragma unroll
    for (int r = 0; r < 8; ++r) {
      int cout = m0 + mtile * 16 + r + 8 * khalf;
      if (cout >= Cout) continue;
      float vo = (j == 0) ? acc0.f[r] : acc1.f[r];
      if (ACT == 1) vo = tanhf(vo);
      yb[(size_t)cout * OHW] = vo;
    }
  }
}

// ---------------------------------------------------------------------------
// Instance norm (+optional affine, ReLU, residual add). In-place safe.
// ---------------------------------------------------------------------------
__global__ __launch_bounds__(256) void inorm_kernel(
    const float* __restrict__ x, const float* __restrict__ g,
    const float* __restrict__ bt, const float* __restrict__ resid,
    float* __restrict__ y, int C, int HW, int relu) {
  const int plane = blockIdx.x;
  const int c = plane % C;
  const float* xp = x + (size_t)plane * HW;
  float s = 0.0f, s2 = 0.0f;
  for (int i = threadIdx.x; i < HW; i += 256) {
    float v = xp[i]; s += v; s2 += v * v;
  }
  __shared__ float r1[256], r2[256];
  r1[threadIdx.x] = s; r2[threadIdx.x] = s2;
  __syncthreads();
  for (int st = 128; st > 0; st >>= 1) {
    if (threadIdx.x < st) {
      r1[threadIdx.x] += r1[threadIdx.x + st];
      r2[threadIdx.x] += r2[threadIdx.x + st];
    }
    __syncthreads();
  }
  __shared__ float mn, inv;
  if (threadIdx.x == 0) {
    float m = r1[0] / (float)HW;
    float var = r2[0] / (float)HW - m * m;
    mn = m; inv = rsqrtf(var + 1e-5f);
  }
  __syncthreads();
  const float ga = g ? g[c] : 1.0f;
  const float be = bt ? bt[c] : 0.0f;
  const float* rp = resid ? (resid + (size_t)plane * HW) : nullptr;
  float* yp = y + (size_t)plane * HW;
  const float lm = mn, li = inv;
  for (int i = threadIdx.x; i < HW; i += 256) {
    float v = (xp[i] - lm) * li * ga + be;
    if (relu) v = fmaxf(v, 0.0f);
    if (rp) v += rp[i];
    yp[i] = v;
  }
}

// ---------------------------------------------------------------------------
// AMM 1x1 convs: og[j]/ob[j] = <fm_r[:,j], gw/bw> + bias
// ---------------------------------------------------------------------------
__global__ void amm_kernel(const float* __restrict__ fmr,
                           const float* __restrict__ gw, const float* __restrict__ gb,
                           const float* __restrict__ bw, const float* __restrict__ bb,
                           float* __restrict__ og, float* __restrict__ ob,
                           int C, int HW, int Bn) {
  int j = blockIdx.x * blockDim.x + threadIdx.x;
  if (j >= Bn * HW) return;
  int b = j / HW, jj = j - b * HW;
  const float* base = fmr + (size_t)b * C * HW + jj;
  float sg = 0.0f, sb = 0.0f;
  for (int c = 0; c < C; ++c) {
    float v = base[(size_t)c * HW];
    sg += v * gw[c]; sb += v * bw[c];
  }
  og[j] = sg + gb[0];
  ob[j] = sb + bb[0];
}

// [B][C][HW] f32 -> [B][HW][C] bf16 with scale (K-contiguous for WMMA loads).
// The 200x attention energy scale is folded into the bf16 quantization of fm_s.
__global__ void transpose_bf16_kernel(const float* __restrict__ fm,
                                      uint16_t* __restrict__ out,
                                      int C, int HW, int Bn, float scale) {
  size_t idx = (size_t)blockIdx.x * blockDim.x + threadIdx.x;
  size_t tot = (size_t)Bn * C * HW;
  if (idx >= tot) return;
  int b = (int)(idx / ((size_t)C * HW));
  size_t rem = idx - (size_t)b * C * HW;
  int c = (int)(rem / HW);
  int i = (int)(rem - (size_t)c * HW);
  out[((size_t)b * HW + i) * C + c] = f2bf(fm[idx] * scale);
}

// ---------------------------------------------------------------------------
// Flash-style token attention:
//   A = softmax_j(200 * <sf[:,i], rf[:,j]>);  gamma[i] = A@og, beta[i] = A@ob
// One wave per 16-token query tile; S via 8 chained bf16 WMMAs (K=256).
// LANE-LOCAL online softmax: each lane owns column (lane&15) of every j-tile
// and keeps its own running (m, l, sum p*og, sum p*ob) -- zero shuffles in the
// hot loop. The 16 per-lane states per row are merged once at the end with 4
// xor rounds of the associative two-state softmax combine.
// ---------------------------------------------------------------------------
__global__ __launch_bounds__(256) void attn_flash_wmma(
    const uint16_t* __restrict__ sfT, const uint16_t* __restrict__ rfT,
    const float* __restrict__ og, const float* __restrict__ ob,
    float* __restrict__ gamma, float* __restrict__ beta, int HW, int C) {
  const int b = blockIdx.y;
  const int wv = threadIdx.x >> 5, lane = threadIdx.x & 31;
  const int ml = lane & 15, khalf = lane >> 4;
  const int i0 = blockIdx.x * 128 + wv * 16;

  FragAB q[8];                                   // full K=256 query fragments
  {
    const uint32_t* qp = (const uint32_t*)(sfT + ((size_t)b * HW + i0 + ml) * C);
#pragma unroll
    for (int kc = 0; kc < 8; ++kc)
#pragma unroll
      for (int v = 0; v < 8; ++v)
        q[kc].u[v] = qp[(kmap(v, khalf) + 32 * kc) >> 1];
  }

  float mrun[8], lrun[8], grun[8], brun[8];
#pragma unroll
  for (int r = 0; r < 8; ++r) { mrun[r] = -3.0e38f; lrun[r] = 0.f; grun[r] = 0.f; brun[r] = 0.f; }

  for (int j0 = 0; j0 < HW; j0 += 16) {
    const uint32_t* kp_ = (const uint32_t*)(rfT + ((size_t)b * HW + j0 + ml) * C);
    FragC S;
#pragma unroll
    for (int i = 0; i < 8; ++i) S.f[i] = 0.0f;
#pragma unroll
    for (int kc = 0; kc < 8; ++kc) {
      FragAB kf;
#pragma unroll
      for (int v = 0; v < 8; ++v) kf.u[v] = kp_[(kmap(v, khalf) + 32 * kc) >> 1];
      S.v = __builtin_amdgcn_wmma_f32_16x16x32_bf16(false, q[kc].v, false, kf.v,
                                                    (short)0, S.v, false, false);
    }
    const float ogv = og[(size_t)b * HW + j0 + ml];
    const float obv = ob[(size_t)b * HW + j0 + ml];
#pragma unroll
    for (int r = 0; r < 8; ++r) {
      float s = S.f[r];                           // 200x folded into sfT
      float mnew = fmaxf(mrun[r], s);
      float sc = __expf(mrun[r] - mnew);          // 1.0 when max unchanged
      float p  = __expf(s - mnew);
      lrun[r] = lrun[r] * sc + p;
      grun[r] = grun[r] * sc + p * ogv;
      brun[r] = brun[r] * sc + p * obv;
      mrun[r] = mnew;
    }
  }

  // merge the 16 per-lane states per row (associative softmax combine)
#pragma unroll
  for (int d = 1; d < 16; d <<= 1) {
#pragma unroll
    for (int r = 0; r < 8; ++r) {
      float mo = __shfl_xor(mrun[r], d, 32);
      float lo = __shfl_xor(lrun[r], d, 32);
      float go = __shfl_xor(grun[r], d, 32);
      float bo = __shfl_xor(brun[r], d, 32);
      float mn = fmaxf(mrun[r], mo);
      float sa = __expf(mrun[r] - mn);
      float sb = __expf(mo - mn);
      lrun[r] = lrun[r] * sa + lo * sb;
      grun[r] = grun[r] * sa + go * sb;
      brun[r] = brun[r] * sa + bo * sb;
      mrun[r] = mn;
    }
  }

  if (ml == 0) {
#pragma unroll
    for (int r = 0; r < 8; ++r) {
      int row = r + 8 * khalf;
      float invl = 1.0f / lrun[r];
      gamma[(size_t)b * HW + i0 + row] = grun[r] * invl;
      beta [(size_t)b * HW + i0 + row] = brun[r] * invl;
    }
  }
}

// morphed = fm_s * (1 + gamma) + beta   (gamma/beta broadcast over C; in-place)
__global__ void morph_kernel(float* __restrict__ fm, const float* __restrict__ gamma,
                             const float* __restrict__ beta, int C, int HW, int Bn) {
  size_t idx = (size_t)blockIdx.x * blockDim.x + threadIdx.x;
  size_t tot = (size_t)Bn * C * HW;
  if (idx >= tot) return;
  int b = (int)(idx / ((size_t)C * HW));
  size_t rem = idx - (size_t)b * C * HW;
  int i = (int)(rem % HW);
  fm[idx] = fm[idx] * (1.0f + gamma[(size_t)b * HW + i]) + beta[(size_t)b * HW + i];
}

// ---------------------------------------------------------------------------
// Host orchestration
// ---------------------------------------------------------------------------
static void launch_conv(const float* x, const uint16_t* A, float* y,
                        int B, int Cin, int H, int W, int Cout, int OH, int OW,
                        int KH, int Kdim, int Kpad, int mode, int act,
                        hipStream_t stream) {
  int Ntot = B * OH * OW;
  dim3 grid((Ntot + 63) / 64, (Cout + 63) / 64);
  if (mode == 1) {
    conv_gemm_wmma<4, 4, 2, 1, 1, 0><<<grid, 256, 0, stream>>>(
        x, A, y, B, Cin, H, W, Cout, OH, OW, Kdim, Kpad);
  } else if (KH == 7 && act == 1) {
    conv_gemm_wmma<7, 7, 1, 3, 0, 1><<<grid, 256, 0, stream>>>(
        x, A, y, B, Cin, H, W, Cout, OH, OW, Kdim, Kpad);
  } else if (KH == 7) {
    conv_gemm_wmma<7, 7, 1, 3, 0, 0><<<grid, 256, 0, stream>>>(
        x, A, y, B, Cin, H, W, Cout, OH, OW, Kdim, Kpad);
  } else if (KH == 4) {
    conv_gemm_wmma<4, 4, 2, 1, 0, 0><<<grid, 256, 0, stream>>>(
        x, A, y, B, Cin, H, W, Cout, OH, OW, Kdim, Kpad);
  } else {
    conv_gemm_wmma<3, 3, 1, 1, 0, 0><<<grid, 256, 0, stream>>>(
        x, A, y, B, Cin, H, W, Cout, OH, OW, Kdim, Kpad);
  }
}

extern "C" void kernel_launch(void* const* d_in, const int* in_sizes, int n_in,
                              void* d_out, int out_size, void* d_ws, size_t ws_size,
                              hipStream_t stream) {
  (void)in_sizes; (void)n_in; (void)out_size; (void)ws_size;
  const int B = 2, HW64 = 64 * 64;

  // Inputs (jax pytree flatten: top-level dict order, then params by sorted keys)
  const float* src = (const float*)d_in[0];
  const float* ref = (const float*)d_in[1];
  // d_in[2], d_in[3]: masks, unused in forward path
  const float* amm_bb = (const float*)d_in[4];
  const float* amm_bw = (const float*)d_in[5];
  const float* amm_gb = (const float*)d_in[6];
  const float* amm_gw = (const float*)d_in[7];
  const float* dec_cf = (const float*)d_in[8];
  const int dec_res_base = 9;   // per res r: +0 b1, +1 b2, +2 g1, +3 g2, +4 w1, +5 w2
  const float* dec_t0  = (const float*)d_in[27];
  const float* dec_t1  = (const float*)d_in[28];
  const float* dec_tb0 = (const float*)d_in[29];
  const float* dec_tb1 = (const float*)d_in[30];
  const float* dec_tg0 = (const float*)d_in[31];
  const float* dec_tg1 = (const float*)d_in[32];
  const float* enc_c0 = (const float*)d_in[33];
  const float* enc_c1 = (const float*)d_in[34];
  const float* enc_c2 = (const float*)d_in[35];
  const int enc_res_base = 36;
  const float* md_b0 = (const float*)d_in[54];
  const float* md_b1 = (const float*)d_in[55];
  const float* md_b2 = (const float*)d_in[56];
  const float* md_c0 = (const float*)d_in[57];
  const float* md_c1 = (const float*)d_in[58];
  const float* md_c2 = (const float*)d_in[59];
  const float* md_g0 = (const float*)d_in[60];
  const float* md_g1 = (const float*)d_in[61];
  const float* md_g2 = (const float*)d_in[62];
  const int md_res_base = 63;

  // workspace bump allocator (256B aligned)
  char* wsp = (char*)d_ws;
  size_t off = 0;
  auto alloc = [&](size_t bytes) -> void* {
    void* p = wsp + off;
    off = (off + bytes + 255) & ~((size_t)255);
    return p;
  };

  auto repack = [&](const float* w, uint16_t* A, int M, int K, int Mpad, int Kpad,
                    int mode, int Cout, int KHW) {
    size_t total = (size_t)Mpad * Kpad;
    repack_w_kernel<<<(unsigned)((total + 255) / 256), 256, 0, stream>>>(
        w, A, M, K, Kpad, total, mode, Cout, KHW);
  };
  auto conv = [&](const float* x, const uint16_t* A, float* y,
                  int Cin, int H, int W, int Cout, int OH, int OW,
                  int KH, int Kdim, int Kpad, int mode, int act) {
    launch_conv(x, A, y, B, Cin, H, W, Cout, OH, OW, KH, Kdim, Kpad, mode, act, stream);
  };
  auto inorm = [&](const float* x, const float* g, const float* bt,
                   const float* resid, float* y, int C, int HWp, int relu) {
    inorm_kernel<<<B * C, 256, 0, stream>>>(x, g, bt, resid, y, C, HWp, relu);
  };

  // bf16 weight arenas
  uint16_t* Wenc0 = (uint16_t*)alloc((size_t)64 * 160 * 2);
  uint16_t* Wenc1 = (uint16_t*)alloc((size_t)128 * 1024 * 2);
  uint16_t* Wenc2 = (uint16_t*)alloc((size_t)256 * 2048 * 2);
  uint16_t* Wmd0  = (uint16_t*)alloc((size_t)64 * 160 * 2);
  uint16_t* Wmd1  = (uint16_t*)alloc((size_t)128 * 1024 * 2);
  uint16_t* Wmd2  = (uint16_t*)alloc((size_t)256 * 2048 * 2);
  uint16_t* WencR[3][2]; uint16_t* WmdR[3][2]; uint16_t* WdecR[3][2];
  for (int r = 0; r < 3; ++r)
    for (int j = 0; j < 2; ++j) {
      WencR[r][j] = (uint16_t*)alloc((size_t)256 * 2304 * 2);
      WmdR[r][j]  = (uint16_t*)alloc((size_t)256 * 2304 * 2);
      WdecR[r][j] = (uint16_t*)alloc((size_t)256 * 2304 * 2);
    }
  uint16_t* Wt0 = (uint16_t*)alloc((size_t)128 * 4096 * 2);
  uint16_t* Wt1 = (uint16_t*)alloc((size_t)64 * 2048 * 2);
  uint16_t* Wcf = (uint16_t*)alloc((size_t)64 * 3136 * 2);

  // activation buffers
  float* ACTA = (float*)alloc((size_t)B * 64 * 256 * 256 * 4);   // 32MB
  float* ACTB = (float*)alloc((size_t)B * 128 * 128 * 128 * 4);  // 16MB
  float* FMS  = (float*)alloc((size_t)B * 256 * HW64 * 4);       // 8MB
  float* FMR  = (float*)alloc((size_t)B * 256 * HW64 * 4);
  float* T1   = (float*)alloc((size_t)B * 256 * HW64 * 4);
  float* T2   = (float*)alloc((size_t)B * 256 * HW64 * 4);
  uint16_t* SFT = (uint16_t*)alloc((size_t)B * HW64 * 256 * 2);  // 4MB bf16
  uint16_t* RFT = (uint16_t*)alloc((size_t)B * HW64 * 256 * 2);
  float* OG    = (float*)alloc((size_t)B * HW64 * 4);
  float* OB    = (float*)alloc((size_t)B * HW64 * 4);
  float* GAMMA = (float*)alloc((size_t)B * HW64 * 4);
  float* BETA  = (float*)alloc((size_t)B * HW64 * 4);

  // ---- weight repack (bf16, zero-padded) ----
  repack(enc_c0, Wenc0, 64, 147, 64, 160, 0, 0, 49);
  repack(enc_c1, Wenc1, 128, 1024, 128, 1024, 0, 0, 16);
  repack(enc_c2, Wenc2, 256, 2048, 256, 2048, 0, 0, 16);
  repack(md_c0, Wmd0, 64, 147, 64, 160, 0, 0, 49);
  repack(md_c1, Wmd1, 128, 1024, 128, 1024, 0, 0, 16);
  repack(md_c2, Wmd2, 256, 2048, 256, 2048, 0, 0, 16);
  for (int r = 0; r < 3; ++r) {
    repack((const float*)d_in[enc_res_base + r * 6 + 4], WencR[r][0], 256, 2304, 256, 2304, 0, 0, 9);
    repack((const float*)d_in[enc_res_base + r * 6 + 5], WencR[r][1], 256, 2304, 256, 2304, 0, 0, 9);
    repack((const float*)d_in[md_res_base + r * 6 + 4], WmdR[r][0], 256, 2304, 256, 2304, 0, 0, 9);
    repack((const float*)d_in[md_res_base + r * 6 + 5], WmdR[r][1], 256, 2304, 256, 2304, 0, 0, 9);
    repack((const float*)d_in[dec_res_base + r * 6 + 4], WdecR[r][0], 256, 2304, 256, 2304, 0, 0, 9);
    repack((const float*)d_in[dec_res_base + r * 6 + 5], WdecR[r][1], 256, 2304, 256, 2304, 0, 0, 9);
  }
  repack(dec_t0, Wt0, 128, 4096, 128, 4096, 1, 128, 16);  // deconv layout (Cin,Cout,4,4)
  repack(dec_t1, Wt1, 64, 2048, 64, 2048, 1, 64, 16);
  repack(dec_cf, Wcf, 3, 3136, 64, 3136, 0, 0, 49);

  auto resblock = [&](float* X, uint16_t* W1, const float* g1, const float* b1,
                      uint16_t* W2, const float* g2, const float* b2) {
    conv(X, W1, T1, 256, 64, 64, 256, 64, 64, 3, 2304, 2304, 0, 0);
    inorm(T1, g1, b1, nullptr, T1, 256, HW64, 1);
    conv(T1, W2, T2, 256, 64, 64, 256, 64, 64, 3, 2304, 2304, 0, 0);
    inorm(T2, g2, b2, X, X, 256, HW64, 0);   // X = X + IN(conv2)
  };

  // ---- encoder (source): INs affine-free ----
  conv(src, Wenc0, ACTA, 3, 256, 256, 64, 256, 256, 7, 147, 160, 0, 0);
  inorm(ACTA, nullptr, nullptr, nullptr, ACTA, 64, 65536, 1);
  conv(ACTA, Wenc1, ACTB, 64, 256, 256, 128, 128, 128, 4, 1024, 1024, 0, 0);
  inorm(ACTB, nullptr, nullptr, nullptr, ACTB, 128, 16384, 1);
  conv(ACTB, Wenc2, FMS, 128, 128, 128, 256, 64, 64, 4, 2048, 2048, 0, 0);
  inorm(FMS, nullptr, nullptr, nullptr, FMS, 256, HW64, 1);
  for (int r = 0; r < 3; ++r)
    resblock(FMS, WencR[r][0], (const float*)d_in[enc_res_base + r * 6 + 2],
             (const float*)d_in[enc_res_base + r * 6 + 0], WencR[r][1],
             (const float*)d_in[enc_res_base + r * 6 + 3],
             (const float*)d_in[enc_res_base + r * 6 + 1]);

  // ---- MDNet (reference): INs affine ----
  conv(ref, Wmd0, ACTA, 3, 256, 256, 64, 256, 256, 7, 147, 160, 0, 0);
  inorm(ACTA, md_g0, md_b0, nullptr, ACTA, 64, 65536, 1);
  conv(ACTA, Wmd1, ACTB, 64, 256, 256, 128, 128, 128, 4, 1024, 1024, 0, 0);
  inorm(ACTB, md_g1, md_b1, nullptr, ACTB, 128, 16384, 1);
  conv(ACTB, Wmd2, FMR, 128, 128, 128, 256, 64, 64, 4, 2048, 2048, 0, 0);
  inorm(FMR, md_g2, md_b2, nullptr, FMR, 256, HW64, 1);
  for (int r = 0; r < 3; ++r)
    resblock(FMR, WmdR[r][0], (const float*)d_in[md_res_base + r * 6 + 2],
             (const float*)d_in[md_res_base + r * 6 + 0], WmdR[r][1],
             (const float*)d_in[md_res_base + r * 6 + 3],
             (const float*)d_in[md_res_base + r * 6 + 1]);

  // ---- AMM attention morph ----
  amm_kernel<<<(B * HW64 + 255) / 256, 256, 0, stream>>>(FMR, amm_gw, amm_gb,
                                                         amm_bw, amm_bb, OG, OB,
                                                         256, HW64, B);
  {
    size_t tot = (size_t)B * 256 * HW64;
    transpose_bf16_kernel<<<(unsigned)((tot + 255) / 256), 256, 0, stream>>>(FMS, SFT, 256, HW64, B, 200.0f);
    transpose_bf16_kernel<<<(unsigned)((tot + 255) / 256), 256, 0, stream>>>(FMR, RFT, 256, HW64, B, 1.0f);
  }
  attn_flash_wmma<<<dim3(HW64 / 128, B), 256, 0, stream>>>(SFT, RFT, OG, OB,
                                                           GAMMA, BETA, HW64, 256);
  {
    size_t tot = (size_t)B * 256 * HW64;
    morph_kernel<<<(unsigned)((tot + 255) / 256), 256, 0, stream>>>(FMS, GAMMA, BETA, 256, HW64, B);
  }

  // ---- decoder ----
  for (int r = 0; r < 3; ++r)
    resblock(FMS, WdecR[r][0], (const float*)d_in[dec_res_base + r * 6 + 2],
             (const float*)d_in[dec_res_base + r * 6 + 0], WdecR[r][1],
             (const float*)d_in[dec_res_base + r * 6 + 3],
             (const float*)d_in[dec_res_base + r * 6 + 1]);
  conv(FMS, Wt0, ACTB, 256, 64, 64, 128, 128, 128, 4, 4096, 4096, 1, 0);
  inorm(ACTB, dec_tg0, dec_tb0, nullptr, ACTB, 128, 16384, 1);
  conv(ACTB, Wt1, ACTA, 128, 128, 128, 64, 256, 256, 4, 2048, 2048, 1, 0);
  inorm(ACTA, dec_tg1, dec_tb1, nullptr, ACTA, 64, 65536, 1);
  conv(ACTA, Wcf, (float*)d_out, 64, 256, 256, 3, 256, 256, 7, 3136, 3136, 0, 1);
}
